// CausalAirEncoderLayer_47330539601994
// MI455X (gfx1250) — matrix operand order repaired and checked by reading, not compiled, not run on hardware
//
#include <hip/hip_runtime.h>
#include <hip/hip_bf16.h>
#include <math.h>

// ---------------- problem constants ----------------
#define PB 8
#define PN 184
#define PC 7
#define PM 56
#define PD 512
#define PH 8
#define PE 64
#define PDF 2048
#define NROWS (PB*PN*PC)        /* 10304 tokens in [B,N,C] order */
#define SCALE 0.125f            /* 1/sqrt(E) */
#define NOISE_STD 0.1f
#define LN_EPS 1e-5f

typedef _Float16 v16h __attribute__((ext_vector_type(16)));
typedef _Float16 h8   __attribute__((ext_vector_type(8)));
typedef float    v8f  __attribute__((ext_vector_type(8)));

union F16x16 { v16h v; h8 p[2]; };

__device__ __forceinline__ h8 pack_h8(const float4& a, const float4& b) {
  h8 r;
  r[0] = (_Float16)a.x; r[1] = (_Float16)a.y;
  r[2] = (_Float16)a.z; r[3] = (_Float16)a.w;
  r[4] = (_Float16)b.x; r[5] = (_Float16)b.y;
  r[6] = (_Float16)b.z; r[7] = (_Float16)b.w;
  return r;
}

// =====================================================================
// GEMM: Y[rows,dout] = act( A[rows,din] @ W[dout,din]^T + bias [+ R] )
// f16 WMMA 16x16x32, f32 accumulate.
// Block 256 thr = 8 waves; block tile 128x128; wave tile 32x64 (2x4 WMMA
// tiles, 8 accumulators). Double-buffered LDS: one barrier per K-tile;
// next K-tile's global loads overlap the current WMMAs. Out-of-range A
// rows are CLAMPED (their products land only in discarded output rows).
// dout must be a multiple of 128.
// =====================================================================
__global__ __launch_bounds__(256)
void gemm_f16(const float* __restrict__ A, const float* __restrict__ W,
              const float* __restrict__ bias, const float* __restrict__ R,
              float* __restrict__ Y, int rows, int din, int dout, int relu)
{
  __shared__ __align__(16) _Float16 As[2][128][40];  // [buf][m][k], 80B rows
  __shared__ __align__(16) _Float16 Bs[2][128][40];  // [buf][n][k] = W[n][k]
  const int tid  = threadIdx.x;
  const int wave = tid >> 5;
  const int lane = tid & 31;
  const int rowBase = blockIdx.x * 128;
  const int colBase = blockIdx.y * 128;
  const int mOff = (wave & 3) * 32;     // 4 waves down
  const int nOff = (wave >> 2) * 64;    // 2 waves across

  v8f acc[2][4] = {};

  const int lr = tid >> 1;              // 0..127 row of tile
  const int lk = (tid & 1) * 16;        // 0 or 16
  const int grA = (rowBase + lr < rows) ? (rowBase + lr) : (rows - 1);
  const float* aRow = A + (size_t)grA * din + lk;
  const float* wRow = W + (size_t)(colBase + lr) * din + lk;

  // prefetch K-tile 0 into registers, stage into buffer 0
  float4 pa0, pa1, pa2, pa3, pw0, pw1, pw2, pw3;
  {
    const float4* ap = (const float4*)aRow;
    pa0 = ap[0]; pa1 = ap[1]; pa2 = ap[2]; pa3 = ap[3];
    const float4* wp = (const float4*)wRow;
    pw0 = wp[0]; pw1 = wp[1]; pw2 = wp[2]; pw3 = wp[3];
  }
  *(h8*)&As[0][lr][lk]     = pack_h8(pa0, pa1);
  *(h8*)&As[0][lr][lk + 8] = pack_h8(pa2, pa3);
  *(h8*)&Bs[0][lr][lk]     = pack_h8(pw0, pw1);
  *(h8*)&Bs[0][lr][lk + 8] = pack_h8(pw2, pw3);

  const int l15 = lane & 15;
  const int aHi = (lane >> 4) * 8;    // A: K groups [hi..hi+7],[hi+16..hi+23]
  const int bHi = (lane >> 4) * 8;    // B: K groups [hi..hi+7],[hi+16..hi+23]

  int cur = 0;
  for (int k0 = 0; k0 < din; k0 += 32) {
    const bool hasNext = (k0 + 32 < din);
    if (hasNext) {   // issue next tile's global loads early (overlap WMMA)
      const float4* ap = (const float4*)(aRow + k0 + 32);
      pa0 = ap[0]; pa1 = ap[1]; pa2 = ap[2]; pa3 = ap[3];
      const float4* wp = (const float4*)(wRow + k0 + 32);
      pw0 = wp[0]; pw1 = wp[1]; pw2 = wp[2]; pw3 = wp[3];
    }
    __syncthreads();   // buffer `cur` fully staged; prior readers of `cur^1` done

    F16x16 afr[2];
    #pragma unroll
    for (int mi = 0; mi < 2; ++mi) {
      const int m = mOff + mi * 16 + l15;
      afr[mi].p[0] = *(const h8*)&As[cur][m][aHi];
      afr[mi].p[1] = *(const h8*)&As[cur][m][aHi + 16];
    }
    #pragma unroll
    for (int ni = 0; ni < 4; ++ni) {
      const int n = nOff + ni * 16 + l15;
      F16x16 bfr;
      bfr.p[0] = *(const h8*)&Bs[cur][n][bHi];
      bfr.p[1] = *(const h8*)&Bs[cur][n][bHi + 16];
      #pragma unroll
      for (int mi = 0; mi < 2; ++mi)
        acc[mi][ni] = __builtin_amdgcn_wmma_f32_16x16x32_f16(
            false, afr[mi].v, false, bfr.v,
            (short)0, acc[mi][ni], false, false);
    }

    if (hasNext) {   // stage next tile into the other buffer (no barrier needed)
      const int nxt = cur ^ 1;
      *(h8*)&As[nxt][lr][lk]     = pack_h8(pa0, pa1);
      *(h8*)&As[nxt][lr][lk + 8] = pack_h8(pa2, pa3);
      *(h8*)&Bs[nxt][lr][lk]     = pack_h8(pw0, pw1);
      *(h8*)&Bs[nxt][lr][lk + 8] = pack_h8(pw2, pw3);
    }
    cur ^= 1;
  }

  // epilogue: C layout — lanes 0-15: N=lane, M=r; lanes 16-31: N=lane-16, M=r+8
  const int rHi = (lane >> 4) * 8;
  const int mBase = rowBase + mOff + rHi;
  if (rowBase + 128 <= rows) {           // fast path: full tile
    #pragma unroll
    for (int ni = 0; ni < 4; ++ni) {
      const int n = colBase + nOff + ni * 16 + l15;
      const float bn = bias[n];
      #pragma unroll
      for (int mi = 0; mi < 2; ++mi) {
        const int m0 = mBase + mi * 16;
        float res[8];
        if (R) {
          #pragma unroll
          for (int r = 0; r < 8; ++r)     // batch residual loads, one wait
            res[r] = R[(size_t)(m0 + r) * dout + n];
        }
        #pragma unroll
        for (int r = 0; r < 8; ++r) {
          float v = acc[mi][ni][r] + bn;
          if (R) v += res[r];
          if (relu) v = fmaxf(v, 0.0f);
          Y[(size_t)(m0 + r) * dout + n] = v;
        }
      }
    }
  } else {                               // guarded slow path (partial tiles)
    #pragma unroll
    for (int ni = 0; ni < 4; ++ni) {
      const int n = colBase + nOff + ni * 16 + l15;
      const float bn = bias[n];
      #pragma unroll
      for (int mi = 0; mi < 2; ++mi) {
        const int m0 = mBase + mi * 16;
        #pragma unroll
        for (int r = 0; r < 8; ++r) {
          const int m = m0 + r;
          if (m < rows) {
            float v = acc[mi][ni][r] + bn;
            if (R) v += R[(size_t)m * dout + n];
            if (relu) v = fmaxf(v, 0.0f);
            Y[(size_t)m * dout + n] = v;
          }
        }
      }
    }
  }
}

// =====================================================================
// noise: hash -> Box-Muller (deterministic stand-in for jax PRNG)
// =====================================================================
__device__ __forceinline__ unsigned hash_u32(unsigned x) {
  x ^= x >> 16; x *= 0x7feb352dU; x ^= x >> 15; x *= 0x846ca68bU; x ^= x >> 16;
  return x;
}
__device__ __forceinline__ void gauss2(unsigned idx, unsigned seed, float* z0, float* z1) {
  unsigned a = hash_u32(idx * 2u + 0x9E3779B9u * (seed + 1u) + 1u);
  unsigned b = hash_u32(idx * 2u + 0x9E3779B9u * (seed + 1u) + 2u);
  float u1 = ((float)a + 1.0f) * 2.3283064e-10f;
  float u2 = (float)b * 2.3283064e-10f;
  float r = sqrtf(-2.0f * __logf(u1));
  float t = 6.2831853f * u2;
  *z0 = r * __cosf(t); *z1 = r * __sinf(t);
}

// =====================================================================
// Attention: one wave per (g, h, l). Row indices parameterized:
//   qRow = (g/qDiv)*qS1 + (g%qDiv)*qS2 + l*qS3
//   kRow = (gk/kDiv)*kS1 + (gk%kDiv)*kS2 + s*kS3, gk = g - kGBase
// maskMode: 0 none | 1 causal[N,N] idx=l*N+s | 2 svc[N,C,M] idx=((g%N)*C+l)*M+s
// backdoor: O = sum_s a*(c*V + (1-c)*Z)
// =====================================================================
__global__ __launch_bounds__(128)
void attn_k(const float* __restrict__ Q, const float* __restrict__ K,
            const float* __restrict__ V, float* __restrict__ AO,
            int G, int gBase, int Lq, int Ls,
            int qDiv, int qS1, int qS2, int qS3,
            int kDiv, int kS1, int kS2, int kS3, int kGBase,
            const float* __restrict__ mask, int maskMode, int seed)
{
  __shared__ float sbuf[4][192];
  const int wave = threadIdx.x >> 5;
  const int lane = threadIdx.x & 31;
  const long wid = (long)blockIdx.x * 4 + wave;
  const long total = (long)G * PH * Lq;
  const bool act = wid < total;

  int l = 0, h = 0, g = 0, qRow = 0, kb = 0;
  float q0 = 0.f, q1 = 0.f;
  if (act) {
    l = (int)(wid % Lq);
    h = (int)((wid / Lq) % PH);
    g = gBase + (int)(wid / ((long)Lq * PH));
    qRow = (g / qDiv) * qS1 + (g % qDiv) * qS2 + l * qS3;
    const int gk = g - kGBase;
    kb = (gk / kDiv) * kS1 + (gk % kDiv) * kS2;
    const float* qp = Q + (size_t)qRow * PD + h * PE + lane * 2;
    q0 = qp[0] * SCALE; q1 = qp[1] * SCALE;
  }

  // pass 1: scores
  float mx = -1e30f;
  if (act) {
    for (int s = 0; s < Ls; ++s) {
      const float* kp = K + (size_t)(kb + s * kS3) * PD + h * PE + lane * 2;
      float p = q0 * kp[0] + q1 * kp[1];
      #pragma unroll
      for (int o = 16; o > 0; o >>= 1) p += __shfl_xor(p, o, 32);
      if (lane == 0) sbuf[wave][s] = p;
      mx = fmaxf(mx, p);
    }
  }
  __syncthreads();

  // pass 2: exp + sum
  float inv = 0.f;
  if (act) {
    float sum = 0.f;
    for (int s = lane; s < Ls; s += 32) {
      float e = __expf(sbuf[wave][s] - mx);
      sbuf[wave][s] = e;
      sum += e;
    }
    #pragma unroll
    for (int o = 16; o > 0; o >>= 1) sum += __shfl_xor(sum, o, 32);
    inv = 1.0f / sum;
  }
  __syncthreads();

  // pass 3: weighted accumulate (+ backdoor noise path)
  if (act) {
    float o0 = 0.f, o1 = 0.f;
    for (int s = 0; s < Ls; ++s) {
      const float a = sbuf[wave][s] * inv;
      const float* vp = V + (size_t)(kb + s * kS3) * PD + h * PE + lane * 2;
      if (maskMode != 0) {
        int mi = (maskMode == 1) ? (l * PN + s)
                                 : (((g % PN) * PC + l) * PM + s);
        const float cm = mask[mi];
        const unsigned zidx =
            ((((unsigned)g * (unsigned)Ls + (unsigned)s) * PH + (unsigned)h) * PE
             + (unsigned)(lane * 2)) >> 1;
        float z0, z1;
        gauss2(zidx, (unsigned)seed, &z0, &z1);
        z0 *= NOISE_STD; z1 *= NOISE_STD;
        o0 += a * (cm * vp[0] + (1.0f - cm) * z0);
        o1 += a * (cm * vp[1] + (1.0f - cm) * z1);
      } else {
        o0 += a * vp[0];
        o1 += a * vp[1];
      }
    }
    float* op = AO + (size_t)qRow * PD + h * PE + lane * 2;
    op[0] = o0; op[1] = o1;
  }
}

// =====================================================================
// LayerNorm over D=512: wave per row (16 elems/lane), optional residual in.
// =====================================================================
__global__ __launch_bounds__(256)
void ln_k(const float* __restrict__ X, const float* __restrict__ Rres,
          const float* __restrict__ gam, const float* __restrict__ bet,
          float* __restrict__ Y, int rows)
{
  const int wave = threadIdx.x >> 5;
  const int lane = threadIdx.x & 31;
  const int row = blockIdx.x * 8 + wave;
  if (row >= rows) return;
  const size_t base = (size_t)row * PD;
  float v[16];
  float s = 0.f;
  #pragma unroll
  for (int j = 0; j < 16; ++j) {
    const int c = lane + j * 32;
    float t = X[base + c];
    if (Rres) t += Rres[base + c];
    v[j] = t; s += t;
  }
  #pragma unroll
  for (int o = 16; o > 0; o >>= 1) s += __shfl_xor(s, o, 32);
  const float mu = s * (1.0f / PD);
  float var = 0.f;
  #pragma unroll
  for (int j = 0; j < 16; ++j) { const float d = v[j] - mu; var += d * d; }
  #pragma unroll
  for (int o = 16; o > 0; o >>= 1) var += __shfl_xor(var, o, 32);
  const float inv = rsqrtf(var * (1.0f / PD) + LN_EPS);
  #pragma unroll
  for (int j = 0; j < 16; ++j) {
    const int c = lane + j * 32;
    Y[base + c] = (v[j] - mu) * inv * gam[c] + bet[c];
  }
}

// =====================================================================
// host orchestration
// =====================================================================
extern "C" void kernel_launch(void* const* d_in, const int* in_sizes, int n_in,
                              void* d_out, int out_size, void* d_ws, size_t ws_size,
                              hipStream_t stream)
{
  const float* aq_x   = (const float*)d_in[0];
  const float* mete_x = (const float*)d_in[1];
  const float* fut_x  = (const float*)d_in[2];
  const float* fmete  = (const float*)d_in[3];
  const float* causal = (const float*)d_in[5];
  const float* svc    = (const float*)d_in[6];
  auto P  = [&](int i) { return (const float*)d_in[i]; };
  // params: 7 + L*8 + {wq,bq,wk,bk,wv,bv,wo,bo}
  // L: ss=0 vs=1 fss=2 fvs=3 hm=4 fm=5 sc=6 vc=7
  auto Wp = [&](int L, int j) { return P(7 + L * 8 + j * 2); };
  auto Bp = [&](int L, int j) { return P(7 + L * 8 + j * 2 + 1); };

  const size_t SZ    = (size_t)NROWS * PD;     // 5,275,648 floats (one tensor)
  const size_t BUFSZ = (size_t)2624 * PDF;     // 5,373,952 floats (max FFN chunk)
  float* ws = (float*)d_ws;
  float* AQ = ws;               float* FU = ws + BUFSZ;
  float* Qb = ws + 2 * BUFSZ;   float* Kb = ws + 3 * BUFSZ;
  float* Vb = ws + 4 * BUFSZ;   float* Ab = ws + 5 * BUFSZ;
  float* KM = ws + 6 * BUFSZ;   float* VM = ws + 7 * BUFSZ;
  float* out = (float*)d_out;

  auto gemm = [&](const float* A, const float* W, const float* b, const float* R,
                  float* Y, int rows, int din, int dout, int relu) {
    dim3 grid((rows + 127) / 128, dout / 128);
    gemm_f16<<<grid, 256, 0, stream>>>(A, W, b, R, Y, rows, din, dout, relu);
  };
  auto attn = [&](const float* Q, const float* K, const float* V, float* AO,
                  int G, int gBase, int Lq, int Ls,
                  int qDiv, int qS1, int qS2, int qS3,
                  int kDiv, int kS1, int kS2, int kS3, int kGBase,
                  const float* mask, int mMode, int seed) {
    long total = (long)G * PH * Lq;
    int blocks = (int)((total + 3) / 4);
    attn_k<<<blocks, 128, 0, stream>>>(Q, K, V, AO, G, gBase, Lq, Ls,
                                       qDiv, qS1, qS2, qS3,
                                       kDiv, kS1, kS2, kS3, kGBase,
                                       mask, mMode, seed);
  };
  auto lnl = [&](const float* X, const float* R, const float* g, const float* b,
                 float* Y, int rows) {
    ln_k<<<(rows + 7) / 8, 256, 0, stream>>>(X, R, g, b, Y, rows);
  };

  const int NC = PN * PC;   // 1288
  // -------- ss: station self-attn on aq_x (causal backdoor, seed 0) --------
  gemm(aq_x, Wp(0,0), Bp(0,0), nullptr, Qb, NROWS, PD, PD, 0);
  gemm(aq_x, Wp(0,1), Bp(0,1), nullptr, Kb, NROWS, PD, PD, 0);
  gemm(aq_x, Wp(0,2), Bp(0,2), nullptr, Vb, NROWS, PD, PD, 0);
  attn(Qb, Kb, Vb, Ab, PB*PC, 0, PN, PN, PC, NC, 1, PC, PC, NC, 1, PC, 0, causal, 1, 0);
  gemm(Ab, Wp(0,3), Bp(0,3), aq_x, AQ, NROWS, PD, PD, 0);     // AQ = aq_x + ss
  // -------- vs: variate self-attn on aq_x (plain) --------
  gemm(aq_x, Wp(1,0), Bp(1,0), nullptr, Qb, NROWS, PD, PD, 0);
  gemm(aq_x, Wp(1,1), Bp(1,1), nullptr, Kb, NROWS, PD, PD, 0);
  gemm(aq_x, Wp(1,2), Bp(1,2), nullptr, Vb, NROWS, PD, PD, 0);
  attn(Qb, Kb, Vb, Ab, PB*PN, 0, PC, PC, 1, PC, 0, 1, 1, PC, 0, 1, 0, nullptr, 0, 0);
  gemm(Ab, Wp(1,3), Bp(1,3), AQ, AQ, NROWS, PD, PD, 0);       // AQ += vs
  // -------- hm: mete cross-attn (svc backdoor, seed 1), chunked K/V --------
  gemm(AQ, Wp(4,0), Bp(4,0), nullptr, Qb, NROWS, PD, PD, 0);
  for (int ci = 0; ci < 8; ++ci) {
    const float* mA = mete_x + (size_t)ci * 184 * PM * PD;
    gemm(mA, Wp(4,1), Bp(4,1), nullptr, KM, 184 * PM, PD, PD, 0);
    gemm(mA, Wp(4,2), Bp(4,2), nullptr, VM, 184 * PM, PD, PD, 0);
    attn(Qb, KM, VM, Ab, 184, ci * 184, PC, PM,
         1, PC, 0, 1, 1, PM, 0, 1, ci * 184, svc, 2, 1);
  }
  gemm(Ab, Wp(4,3), Bp(4,3), AQ, AQ, NROWS, PD, PD, 0);       // AQ += hm
  // -------- sc: station cross future->hist (causal backdoor, seed 2) --------
  gemm(fut_x, Wp(6,0), Bp(6,0), nullptr, Qb, NROWS, PD, PD, 0);
  gemm(AQ,    Wp(6,1), Bp(6,1), nullptr, Kb, NROWS, PD, PD, 0);
  gemm(AQ,    Wp(6,2), Bp(6,2), nullptr, Vb, NROWS, PD, PD, 0);
  attn(Qb, Kb, Vb, Ab, PB*PC, 0, PN, PN, PC, NC, 1, PC, PC, NC, 1, PC, 0, causal, 1, 2);
  gemm(Ab, Wp(6,3), Bp(6,3), fut_x, FU, NROWS, PD, PD, 0);    // FU = fut_x + sc
  // -------- vc: variate cross (plain) --------
  gemm(fut_x, Wp(7,0), Bp(7,0), nullptr, Qb, NROWS, PD, PD, 0);
  gemm(AQ,    Wp(7,1), Bp(7,1), nullptr, Kb, NROWS, PD, PD, 0);
  gemm(AQ,    Wp(7,2), Bp(7,2), nullptr, Vb, NROWS, PD, PD, 0);
  attn(Qb, Kb, Vb, Ab, PB*PN, 0, PC, PC, 1, PC, 0, 1, 1, PC, 0, 1, 0, nullptr, 0, 0);
  gemm(Ab, Wp(7,3), Bp(7,3), FU, FU, NROWS, PD, PD, 0);       // FU += vc
  // -------- fm: fore mete cross-attn (svc backdoor, seed 3), chunked --------
  gemm(FU, Wp(5,0), Bp(5,0), nullptr, Qb, NROWS, PD, PD, 0);
  for (int ci = 0; ci < 8; ++ci) {
    const float* mA = fmete + (size_t)ci * 184 * PM * PD;
    gemm(mA, Wp(5,1), Bp(5,1), nullptr, KM, 184 * PM, PD, PD, 0);
    gemm(mA, Wp(5,2), Bp(5,2), nullptr, VM, 184 * PM, PD, PD, 0);
    attn(Qb, KM, VM, Ab, 184, ci * 184, PC, PM,
         1, PC, 0, 1, 1, PM, 0, 1, ci * 184, svc, 2, 3);
  }
  gemm(Ab, Wp(5,3), Bp(5,3), FU, FU, NROWS, PD, PD, 0);       // FU += fm
  // -------- fss (causal backdoor, seed 4) + fvs (plain), both from FU --------
  gemm(FU, Wp(2,0), Bp(2,0), nullptr, Qb, NROWS, PD, PD, 0);
  gemm(FU, Wp(2,1), Bp(2,1), nullptr, Kb, NROWS, PD, PD, 0);
  gemm(FU, Wp(2,2), Bp(2,2), nullptr, Vb, NROWS, PD, PD, 0);
  attn(Qb, Kb, Vb, Ab, PB*PC, 0, PN, PN, PC, NC, 1, PC, PC, NC, 1, PC, 0, causal, 1, 4);
  gemm(FU, Wp(3,0), Bp(3,0), nullptr, Qb, NROWS, PD, PD, 0);
  gemm(FU, Wp(3,1), Bp(3,1), nullptr, Kb, NROWS, PD, PD, 0);
  gemm(FU, Wp(3,2), Bp(3,2), nullptr, Vb, NROWS, PD, PD, 0);
  attn(Qb, Kb, Vb, KM, PB*PN, 0, PC, PC, 1, PC, 0, 1, 1, PC, 0, 1, 0, nullptr, 0, 0);
  gemm(Ab, Wp(2,3), Bp(2,3), FU, VM, NROWS, PD, PD, 0);       // VM = FU + fss
  gemm(KM, Wp(3,3), Bp(3,3), VM, FU, NROWS, PD, PD, 0);       // FU = VM + fvs
  // -------- FFN (aq): ln1 -> relu(conv1) -> conv2 + xn -> ln2 --------
  static const int chunks[4] = {2624, 2624, 2624, 2432};      // sum = 10304
  lnl(AQ, nullptr, P(75), P(76), Qb, NROWS);                  // xn in Qb
  {
    int r0 = 0;
    for (int c = 0; c < 4; ++c) {
      const int nr = chunks[c];
      gemm(Qb + (size_t)r0 * PD, P(71), P(72), nullptr, Vb, nr, PD, PDF, 1);
      gemm(Vb, P(73), P(74), Qb + (size_t)r0 * PD, Kb + (size_t)r0 * PD, nr, PDF, PD, 0);
      r0 += nr;
    }
  }
  lnl(Kb, nullptr, P(77), P(78), out, NROWS);                 // out[0] = ffn(aq)
  // -------- FFN (future) --------
  lnl(FU, nullptr, P(83), P(84), Qb, NROWS);
  {
    int r0 = 0;
    for (int c = 0; c < 4; ++c) {
      const int nr = chunks[c];
      gemm(Qb + (size_t)r0 * PD, P(79), P(80), nullptr, Vb, nr, PD, PDF, 1);
      gemm(Vb, P(81), P(82), Qb + (size_t)r0 * PD, Kb + (size_t)r0 * PD, nr, PDF, PD, 0);
      r0 += nr;
    }
  }
  lnl(Kb, nullptr, P(85), P(86), out + SZ, NROWS);            // out[1] = ffn(fut)
}
